// CustomGPT2Attention_27479200760085
// MI455X (gfx1250) — matrix operand order repaired and checked by reading, not compiled, not run on hardware
//
#include <hip/hip_runtime.h>
#include <hip/hip_bf16.h>

// ---------------- problem constants (from reference) ----------------
#define BB 4
#define NN 4096
#define EE 65536
#define DD 768
#define HH 12
#define CC 64
#define ETOT (EE + NN)          // edges + self loops = 69632
#define NEG_SLOPE 0.2f

typedef __bf16 bf16_t;
typedef __attribute__((ext_vector_type(16))) __bf16 v16bf;
typedef __attribute__((ext_vector_type(8)))  float  v8f;

// ---------------- GEMM: xh = X @ W  (bf16 WMMA, fp32 accum) ----------------
// Block tile 128(M) x 64(N), K-tile 64, double-buffered LDS, 8 WMMA / barrier.
#define MT 128
#define NT 64
#define KTILE 64
#define LDA 68   // padded bf16 stride (136B): 16-row column reads conflict-free
#define LDB 68

// A fragment (16x32 bf16), ISA 7.12.2 layout:
// lane l: row M = base + (l&15); group g = l>>4
// element j (0..15): k = (j<8 ? 0 : 16) + g*8 + (j&7)  -> 32-bit pair loads
__device__ __forceinline__ v16bf load_a_frag(const bf16_t* Alds, int row_base,
                                             int ks, int lane) {
    union { unsigned int u[8]; v16bf v; } r;
    const int m = row_base + (lane & 15);
    const int g = (lane >> 4);
    const unsigned int* rowp = (const unsigned int*)(Alds + m * LDA + ks);
#pragma unroll
    for (int p = 0; p < 8; ++p) {
        const int k = ((p < 4) ? 0 : 16) + g * 8 + (p & 3) * 2;
        r.u[p] = rowp[k >> 1];
    }
    return r.v;
}

// B fragment (32x16 bf16): lane l: col N = base + (l&15); g = l>>4
// element j: k = (j<8 ? 0 : 16) + g*8 + (j&7); B staged transposed in LDS: Bt[n][k]
__device__ __forceinline__ v16bf load_b_frag(const bf16_t* Bt, int col_base,
                                             int ks, int lane) {
    union { unsigned int u[8]; v16bf v; } r;
    const int n = col_base + (lane & 15);
    const int g = (lane >> 4);
    const unsigned int* p = (const unsigned int*)(Bt + n * LDB + ks);
#pragma unroll
    for (int q = 0; q < 8; ++q) r.u[q] = p[((q < 4) ? 0 : 8) + g * 4 + (q & 3)];
    return r.v;
}

__global__ __launch_bounds__(256) void gat_gemm_kernel(const float* __restrict__ X,
                                                       const float* __restrict__ W,
                                                       float* __restrict__ XH) {
    __shared__ bf16_t Alds[2][MT * LDA];
    __shared__ bf16_t Blds[2][NT * LDB];   // transposed: [n][k]

    const int tid  = threadIdx.x;
    const int lane = tid & 31;
    const int wave = tid >> 5;
    const int m0 = blockIdx.y * MT;     // rows over B*N = 16384
    const int n0 = blockIdx.x * NT;     // cols over 768
    const int mw = (wave >> 1) * 32;    // wave sub-tile origin
    const int nw = (wave & 1) * 32;

    v8f acc[2][2] = {};
    float4 areg[8];
    float4 breg[4];

    // staging indices
    const int arow = tid >> 1;          // 0..127
    const int acb  = (tid & 1) * 32;    // 0 / 32
    const int brow = tid >> 2;          // k row 0..63
    const int bcb  = (tid & 3) * 16;    // col base

    auto load_next = [&](int k0) {
        const float4* ap = (const float4*)(X + (size_t)(m0 + arow) * DD + k0 + acb);
#pragma unroll
        for (int i = 0; i < 8; ++i) areg[i] = ap[i];
        const float4* bp = (const float4*)(W + (size_t)(k0 + brow) * DD + n0 + bcb);
#pragma unroll
        for (int i = 0; i < 4; ++i) breg[i] = bp[i];
    };
    auto store_tile = [&](bf16_t* Ab, bf16_t* Bb) {
        bf16_t* ad = Ab + arow * LDA + acb;
#pragma unroll
        for (int i = 0; i < 8; ++i) {
            float4 f = areg[i];
            ad[i * 4 + 0] = (bf16_t)f.x;
            ad[i * 4 + 1] = (bf16_t)f.y;
            ad[i * 4 + 2] = (bf16_t)f.z;
            ad[i * 4 + 3] = (bf16_t)f.w;
        }
#pragma unroll
        for (int i = 0; i < 4; ++i) {
            float4 f = breg[i];
            const int c = bcb + i * 4;
            Bb[(c + 0) * LDB + brow] = (bf16_t)f.x;
            Bb[(c + 1) * LDB + brow] = (bf16_t)f.y;
            Bb[(c + 2) * LDB + brow] = (bf16_t)f.z;
            Bb[(c + 3) * LDB + brow] = (bf16_t)f.w;
        }
    };

    // prologue: stage K-tile 0 into buffer 0
    load_next(0);
    store_tile(Alds[0], Blds[0]);
    __syncthreads();

    const int NK = DD / KTILE;   // 12
    for (int kt = 0; kt < NK; ++kt) {
        const int cur = kt & 1;
        if (kt + 1 < NK) load_next((kt + 1) * KTILE);   // overlap with WMMA below

#pragma unroll
        for (int ks = 0; ks < KTILE; ks += 32) {
            v16bf a0 = load_a_frag(Alds[cur], mw + 0,  ks, lane);
            v16bf a1 = load_a_frag(Alds[cur], mw + 16, ks, lane);
            v16bf b0 = load_b_frag(Blds[cur], nw + 0,  ks, lane);
            v16bf b1 = load_b_frag(Blds[cur], nw + 16, ks, lane);
            acc[0][0] = __builtin_amdgcn_wmma_f32_16x16x32_bf16(false, a0, false, b0, (short)0, acc[0][0], false, false);
            acc[0][1] = __builtin_amdgcn_wmma_f32_16x16x32_bf16(false, a0, false, b1, (short)0, acc[0][1], false, false);
            acc[1][0] = __builtin_amdgcn_wmma_f32_16x16x32_bf16(false, a1, false, b0, (short)0, acc[1][0], false, false);
            acc[1][1] = __builtin_amdgcn_wmma_f32_16x16x32_bf16(false, a1, false, b1, (short)0, acc[1][1], false, false);
        }

        if (kt + 1 < NK) store_tile(Alds[1 - cur], Blds[1 - cur]);
        __syncthreads();
    }

    // D layout: lane l, vgpr r -> row = r + 8*(l>>4), col = l&15
#pragma unroll
    for (int mt = 0; mt < 2; ++mt)
#pragma unroll
        for (int nt = 0; nt < 2; ++nt) {
            const int colg  = n0 + nw + nt * 16 + (lane & 15);
            const int rowg0 = m0 + mw + mt * 16 + ((lane >> 4) << 3);
#pragma unroll
            for (int r = 0; r < 8; ++r)
                XH[(size_t)(rowg0 + r) * DD + colg] = acc[mt][nt][r];
        }
}

// ---------------- per-node attention coefficients (wave32 per (b,n,h)) ----------------
__global__ void gat_attn_kernel(const float* __restrict__ XH,
                                const float* __restrict__ att_src,
                                const float* __restrict__ att_dst,
                                float* __restrict__ as_, float* __restrict__ ad_) {
    const int gw = blockIdx.x * (blockDim.x >> 5) + (threadIdx.x >> 5); // wave id over B*N*H
    if (gw >= BB * NN * HH) return;
    const int lane = threadIdx.x & 31;
    const int h = gw % HH;
    const float* xp = XH + (size_t)gw * CC;   // == (b*N+n)*768 + h*64
    const float v0 = xp[lane], v1 = xp[lane + 32];
    float accs = v0 * att_src[h * CC + lane] + v1 * att_src[h * CC + lane + 32];
    float accd = v0 * att_dst[h * CC + lane] + v1 * att_dst[h * CC + lane + 32];
#pragma unroll
    for (int off = 16; off; off >>= 1) {
        accs += __shfl_xor(accs, off, 32);
        accd += __shfl_xor(accd, off, 32);
    }
    if (lane == 0) { as_[gw] = accs; ad_[gw] = accd; }
}

// ---------------- init: out=0 (vectorized), m=-inf, denom=0 ----------------
__global__ void gat_init_kernel(float4* __restrict__ out4, float* __restrict__ m,
                                float* __restrict__ denom) {
    const int idx = blockIdx.x * blockDim.x + threadIdx.x;
    if (idx < BB * NN * (DD / 4)) out4[idx] = make_float4(0.f, 0.f, 0.f, 0.f);
    if (idx < BB * NN * HH) { m[idx] = -3.0e38f; denom[idx] = 0.f; }
}

__device__ __forceinline__ void atomicMaxFloat(float* addr, float val) {
    if (val >= 0.f) atomicMax((int*)addr, __float_as_int(val));
    else            atomicMin((unsigned int*)addr, __float_as_uint(val));
}

// ---------------- edge logits + segment max ----------------
__global__ void gat_edge_logit_kernel(const int* __restrict__ ei,
                                      const float* __restrict__ as_,
                                      const float* __restrict__ ad_,
                                      float* __restrict__ ebuf,
                                      float* __restrict__ m) {
    const int idx = blockIdx.x * blockDim.x + threadIdx.x;   // over B*ETOT*H
    if (idx >= BB * ETOT * HH) return;
    const int h = idx % HH;
    const int t = idx / HH;
    const int k = t % ETOT;
    const int b = t / ETOT;
    const int s = (k < EE) ? ei[k]      : (k - EE);
    const int d = (k < EE) ? ei[EE + k] : (k - EE);
    float e = as_[(b * NN + s) * HH + h] + ad_[(b * NN + d) * HH + h];
    e = (e > 0.f) ? e : e * NEG_SLOPE;
    ebuf[idx] = e;
    atomicMaxFloat(&m[(b * NN + d) * HH + h], e);
}

// ---------------- exp(e - m) + segment sum ----------------
__global__ void gat_edge_exp_kernel(const int* __restrict__ ei,
                                    const float* __restrict__ m,
                                    float* __restrict__ ebuf,
                                    float* __restrict__ denom) {
    const int idx = blockIdx.x * blockDim.x + threadIdx.x;   // over B*ETOT*H
    if (idx >= BB * ETOT * HH) return;
    const int h = idx % HH;
    const int t = idx / HH;
    const int k = t % ETOT;
    const int b = t / ETOT;
    const int d = (k < EE) ? ei[EE + k] : (k - EE);
    const float w = __expf(ebuf[idx] - m[(b * NN + d) * HH + h]);
    ebuf[idx] = w;
    atomicAdd(&denom[(b * NN + d) * HH + h], w);
}

// ---------------- scatter-aggregate: out[dst] += w * xh[src] (4 cols/thread) ----------------
__global__ void gat_aggregate_kernel(const int* __restrict__ ei,
                                     const float* __restrict__ wbuf,
                                     const float* __restrict__ XH,
                                     float* __restrict__ out) {
    const long long idx = (long long)blockIdx.x * blockDim.x + threadIdx.x;
    if (idx >= (long long)BB * ETOT * (DD / 4)) return;
    const int q = (int)(idx % (DD / 4));      // 0..191
    const long long t = idx / (DD / 4);
    const int k = (int)(t % ETOT);
    const int b = (int)(t / ETOT);
    const int s = (k < EE) ? ei[k]      : (k - EE);
    const int d = (k < EE) ? ei[EE + k] : (k - EE);
    const int col = q * 4;
    const float w = wbuf[((long long)b * ETOT + k) * HH + (col >> 6)];
    const float4 xv = *(const float4*)(XH + ((size_t)(b * NN + s)) * DD + col);
    float* op = out + ((size_t)(b * NN + d)) * DD + col;
    atomicAdd(op + 0, w * xv.x);
    atomicAdd(op + 1, w * xv.y);
    atomicAdd(op + 2, w * xv.z);
    atomicAdd(op + 3, w * xv.w);
}

// ---------------- finalize: normalize + bias (4 cols/thread) ----------------
__global__ void gat_finalize_kernel(float* __restrict__ out,
                                    const float* __restrict__ denom,
                                    const float* __restrict__ bias) {
    const int idx = blockIdx.x * blockDim.x + threadIdx.x;   // over B*N*192
    if (idx >= BB * NN * (DD / 4)) return;
    const int q  = idx % (DD / 4);
    const int bn = idx / (DD / 4);
    const int col = q * 4;
    const float inv = 1.f / (denom[bn * HH + (col >> 6)] + 1e-16f);
    float4 o = *(float4*)(out + (size_t)bn * DD + col);
    const float4 bv = *(const float4*)(bias + col);
    o.x = o.x * inv + bv.x;
    o.y = o.y * inv + bv.y;
    o.z = o.z * inv + bv.z;
    o.w = o.w * inv + bv.w;
    *(float4*)(out + (size_t)bn * DD + col) = o;
}

// ---------------- host launcher ----------------
extern "C" void kernel_launch(void* const* d_in, const int* in_sizes, int n_in,
                              void* d_out, int out_size, void* d_ws, size_t ws_size,
                              hipStream_t stream) {
    const float* x       = (const float*)d_in[0];  // [B,N,D]
    const int*   ei      = (const int*)d_in[1];    // [2,E]
    const float* W       = (const float*)d_in[2];  // [D, H*C]
    const float* att_src = (const float*)d_in[3];  // [H*C]
    const float* att_dst = (const float*)d_in[4];  // [H*C]
    const float* bias    = (const float*)d_in[5];  // [H*C]
    float* out = (float*)d_out;

    // workspace partition (floats)
    float* ws    = (float*)d_ws;
    float* xh    = ws;                                   // B*N*768
    float* as_   = xh  + (size_t)BB * NN * DD;           // B*N*H
    float* ad_   = as_ + (size_t)BB * NN * HH;           // B*N*H
    float* m     = ad_ + (size_t)BB * NN * HH;           // B*N*H
    float* denom = m   + (size_t)BB * NN * HH;           // B*N*H
    float* ebuf  = denom + (size_t)BB * NN * HH;         // B*ETOT*H

    // 1) GEMM xh = x @ W  (WMMA bf16 -> fp32, double-buffered LDS)
    {
        dim3 grid(DD / NT, (BB * NN) / MT);   // (12, 128)
        gat_gemm_kernel<<<grid, 256, 0, stream>>>(x, W, xh);
    }
    // 2) per-node attention coefficients (wave32 per (b,n,h))
    {
        const int waves = BB * NN * HH;                  // 196608
        const int blocks = (waves + 7) / 8;              // 8 waves per 256-thread block
        gat_attn_kernel<<<blocks, 256, 0, stream>>>(xh, att_src, att_dst, as_, ad_);
    }
    // 3) init out / m / denom
    {
        const int total = BB * NN * (DD / 4);
        gat_init_kernel<<<(total + 255) / 256, 256, 0, stream>>>((float4*)out, m, denom);
    }
    // 4) edge logits + segment max
    {
        const int total = BB * ETOT * HH;
        gat_edge_logit_kernel<<<(total + 255) / 256, 256, 0, stream>>>(ei, as_, ad_, ebuf, m);
    }
    // 5) exp + segment sum
    {
        const int total = BB * ETOT * HH;
        gat_edge_exp_kernel<<<(total + 255) / 256, 256, 0, stream>>>(ei, m, ebuf, denom);
    }
    // 6) scatter-aggregate (float4 per thread)
    {
        const long long total = (long long)BB * ETOT * (DD / 4);
        const long long blocks = (total + 255) / 256;
        gat_aggregate_kernel<<<(unsigned)blocks, 256, 0, stream>>>(ei, ebuf, xh, out);
    }
    // 7) finalize
    {
        const int total = BB * NN * (DD / 4);
        gat_finalize_kernel<<<(total + 255) / 256, 256, 0, stream>>>(out, denom, bias);
    }
}
